// LSTM_23596550324298
// MI455X (gfx1250) — compile-verified, matrix operand order checked by reading, not run on hardware
//
#include <hip/hip_runtime.h>

#define T_STEPS 2048
#define BATCH   64
#define HID     512
#define NWG     16   // workgroups per layer

typedef __attribute__((ext_vector_type(16))) __bf16 v16bf;
typedef __attribute__((ext_vector_type(8)))  float  v8f;

union Frag { v16bf v; uint4 q[2]; };

__device__ __forceinline__ unsigned short f2bf(float f) {
  unsigned u = __float_as_uint(f);
  u += 0x7FFFu + ((u >> 16) & 1u);           // round-to-nearest-even
  return (unsigned short)(u >> 16);
}

__device__ __forceinline__ float sigmoidf(float x) {
  return 1.0f / (1.0f + __expf(-x));
}

// ---------------------------------------------------------------------------
// Workspace zero-init (h double buffers + barrier counters)
// ---------------------------------------------------------------------------
__global__ void init_ws(unsigned short* hbuf, unsigned int* ctr) {
  int idx = blockIdx.x * blockDim.x + threadIdx.x;
  if (idx < 2 * 2 * BATCH * HID) hbuf[idx] = 0;
  if (idx < 2) ctr[idx] = 0;
}

// ---------------------------------------------------------------------------
// fp32 -> bf16 weight conversion
// ---------------------------------------------------------------------------
__global__ void cvt_bf16(const float* __restrict__ s,
                         unsigned short* __restrict__ d, int n) {
  int i = (blockIdx.x * blockDim.x + threadIdx.x) * 4;
  if (i + 3 < n) {
    float4 f = *(const float4*)(s + i);
    d[i]     = f2bf(f.x);
    d[i + 1] = f2bf(f.y);
    d[i + 2] = f2bf(f.z);
    d[i + 3] = f2bf(f.w);
  }
}

// ---------------------------------------------------------------------------
// Persistent 2-layer LSTM. Grid = 32 blocks x 256 threads.
//   blocks 0..15  -> layer 0 (hidden slice of 32 units each)
//   blocks 16..31 -> layer 1 (pipelined behind layer 0 via ctr[0])
// Each wave owns a 16-column strip of the combined [x|h] x [Wih|Whh]^T GEMM,
// with its bf16 weight strip held in 256 VGPRs for the entire sequence.
// ---------------------------------------------------------------------------
__global__ __launch_bounds__(256, 1) void lstm_persistent(
    const float* __restrict__ x,            // [B,T,D] fp32
    const unsigned short* __restrict__ wih0,
    const unsigned short* __restrict__ whh0,
    const unsigned short* __restrict__ wih1,
    const unsigned short* __restrict__ whh1,
    const float* __restrict__ bih0, const float* __restrict__ bhh0,
    const float* __restrict__ bih1, const float* __restrict__ bhh1,
    unsigned short* __restrict__ hbuf,      // [2 layers][2 bufs][B*H] bf16
    unsigned short* __restrict__ out0,      // [T,B,H] bf16 (layer0 output)
    float* __restrict__ out,                // d_out
    unsigned int* __restrict__ ctr)         // [2] barrier counters
{
  const int l   = blockIdx.x >> 4;          // layer
  const int wg  = blockIdx.x & 15;          // hidden-slice id (32 units)
  const int tid = threadIdx.x;
  const int w   = tid >> 5;                 // wave 0..7
  const int lane = tid & 31;

  const unsigned short* Wih = l ? wih1 : wih0;
  const unsigned short* Whh = l ? whh1 : whh0;
  const float* bi = l ? bih1 : bih0;
  const float* bh = l ? bhh1 : bhh0;
  unsigned short* hb = hbuf + (size_t)l * 2 * BATCH * HID;

  const size_t HN_BASE = (size_t)BATCH * T_STEPS * HID;       // 67108864
  const size_t CN_BASE = HN_BASE + (size_t)2 * BATCH * HID;

  // LDS: A-staging (x_t | h) bf16, padded rows; gate scratch; cell state.
  __shared__ alignas(16) unsigned short Abuf[BATCH][1032];   // 132 KB
  __shared__ float gates[BATCH][132];                        // ~33.8 KB (padded)
  __shared__ float cst[BATCH][32];                           // 8 KB
  __shared__ float bsum[4][32];                              // bias sums

  // zero cell state
  #pragma unroll
  for (int u = 0; u < 8; ++u) {
    int p = tid * 8 + u;
    cst[p >> 5][p & 31] = 0.0f;
  }
  // precompute bias sums for this WG's 128 gate rows
  if (tid < 128) {
    int gi = tid >> 5, j = tid & 31;
    int r  = gi * HID + wg * 32 + j;
    bsum[gi][j] = bi[r] + bh[r];
  }

  // ---- Preload this wave's weight strip into VGPRs (stays for all T) ----
  const int nloc = w * 16 + (lane & 15);       // 0..127 within WG slice
  const int gi0  = nloc >> 5;                  // gate of this column
  const int j0   = nloc & 31;
  const int wrow = gi0 * HID + wg * 32 + j0;   // row in [4H, 512] weight
  const int koff = (lane >> 4) ? 8 : 0;        // lane K-chunk offset

  Frag bF[32];
  #pragma unroll
  for (int ki = 0; ki < 32; ++ki) {
    const unsigned short* src = (ki < 16) ? Wih : Whh;
    int kb = (ki & 15) * 32 + koff;
    bF[ki].q[0] = *(const uint4*)(src + (size_t)wrow * 512 + kb);
    bF[ki].q[1] = *(const uint4*)(src + (size_t)wrow * 512 + kb + 16);
  }
  __syncthreads();

  for (int t = 0; t < T_STEPS; ++t) {
    // -------- layer-1 waits for layer-0 step t to be published --------
    if (l == 1) {
      if (tid == 0) {
        unsigned target = (unsigned)(NWG * (t + 1));
        while (__hip_atomic_load(&ctr[0], __ATOMIC_ACQUIRE,
                                 __HIP_MEMORY_SCOPE_AGENT) < target) {
          __builtin_amdgcn_s_sleep(1);
        }
      }
      __syncthreads();
    }

    // -------- stage A = [x_t | h_{t-1}] into LDS (bf16) --------
    if (l == 0) {
      // x part: fp32 -> bf16, 8192 float4
      #pragma unroll 4
      for (int it = 0; it < 32; ++it) {
        int idx = tid + it * 256;
        int row = idx >> 7;                  // batch
        int c4  = idx & 127;
        float4 f = *(const float4*)(x + ((size_t)row * T_STEPS + t) * 512 + c4 * 4);
        uint2 v;
        v.x = (unsigned)f2bf(f.x) | ((unsigned)f2bf(f.y) << 16);
        v.y = (unsigned)f2bf(f.z) | ((unsigned)f2bf(f.w) << 16);
        *(uint2*)&Abuf[row][c4 * 4] = v;
      }
      // prefetch next step's x slice into L2 (fire-and-forget, hides HBM
      // latency on the serial critical path). 1024 cachelines of 128B.
      if (t + 1 < T_STEPS) {
        #pragma unroll
        for (int k = 0; k < 4; ++k) {
          int li  = tid + k * 256;           // line index 0..1023
          int row = li >> 4;                 // 16 lines per row
          int off = (li & 15) * 32;          // floats
          __builtin_prefetch(x + ((size_t)row * T_STEPS + (t + 1)) * 512 + off, 0, 0);
        }
      }
    } else {
      // x part: layer0 bf16 output (L2-resident, written ~1 step ago)
      #pragma unroll 4
      for (int it = 0; it < 16; ++it) {
        int idx = tid + it * 256;
        int row = idx >> 6;
        int q   = idx & 63;
        uint4 v = *(const uint4*)(out0 + ((size_t)t * BATCH + row) * 512 + q * 8);
        *(uint4*)&Abuf[row][q * 8] = v;
      }
    }
    {
      // h part (bf16 double buffer)
      const unsigned short* hsrc = hb + (size_t)(t & 1) * BATCH * HID;
      #pragma unroll 4
      for (int it = 0; it < 16; ++it) {
        int idx = tid + it * 256;
        int row = idx >> 6;
        int q   = idx & 63;
        uint4 v = *(const uint4*)(hsrc + (size_t)row * 512 + q * 8);
        *(uint4*)&Abuf[row][512 + q * 8] = v;
      }
    }
    __syncthreads();

    // -------- GEMM: gates[64 x 128slice] += A[64x1024] * W^T --------
    v8f acc[4];
    #pragma unroll
    for (int mt = 0; mt < 4; ++mt)
      #pragma unroll
      for (int i = 0; i < 8; ++i) acc[mt][i] = 0.0f;

    #pragma unroll
    for (int ki = 0; ki < 32; ++ki) {
      int kb = ki * 32 + koff;
      int r0 = lane & 15;
      // load all four M-tile fragments first (8 ds_load_b128 in flight,
      // single dscnt wait), then issue the four WMMAs back-to-back.
      Frag a0, a1, a2, a3;
      a0.q[0] = *(const uint4*)&Abuf[r0][kb];
      a0.q[1] = *(const uint4*)&Abuf[r0][kb + 16];
      a1.q[0] = *(const uint4*)&Abuf[16 + r0][kb];
      a1.q[1] = *(const uint4*)&Abuf[16 + r0][kb + 16];
      a2.q[0] = *(const uint4*)&Abuf[32 + r0][kb];
      a2.q[1] = *(const uint4*)&Abuf[32 + r0][kb + 16];
      a3.q[0] = *(const uint4*)&Abuf[48 + r0][kb];
      a3.q[1] = *(const uint4*)&Abuf[48 + r0][kb + 16];
      acc[0] = __builtin_amdgcn_wmma_f32_16x16x32_bf16(
          false, a0.v, false, bF[ki].v, (short)0, acc[0], false, false);
      acc[1] = __builtin_amdgcn_wmma_f32_16x16x32_bf16(
          false, a1.v, false, bF[ki].v, (short)0, acc[1], false, false);
      acc[2] = __builtin_amdgcn_wmma_f32_16x16x32_bf16(
          false, a2.v, false, bF[ki].v, (short)0, acc[2], false, false);
      acc[3] = __builtin_amdgcn_wmma_f32_16x16x32_bf16(
          false, a3.v, false, bF[ki].v, (short)0, acc[3], false, false);
    }

    // -------- scatter accumulators to LDS gate scratch --------
    {
      int col   = w * 16 + (lane & 15);
      int rbase = (lane >> 4) ? 8 : 0;
      #pragma unroll
      for (int mt = 0; mt < 4; ++mt)
        #pragma unroll
        for (int i = 0; i < 8; ++i)
          gates[mt * 16 + rbase + i][col] = acc[mt][i];
    }
    __syncthreads();

    // -------- pointwise LSTM cell: 2048 (b,j) pairs, 8 per thread --------
    {
      unsigned short* hdst = hb + (size_t)((t + 1) & 1) * BATCH * HID;
      #pragma unroll
      for (int u = 0; u < 8; ++u) {
        int p = tid * 8 + u;
        int b = p >> 5;
        int j = p & 31;
        int hidx = wg * 32 + j;
        float ig = gates[b][j]      + bsum[0][j];
        float fg = gates[b][32 + j] + bsum[1][j];
        float gg = gates[b][64 + j] + bsum[2][j];
        float og = gates[b][96 + j] + bsum[3][j];
        float i_ = sigmoidf(ig);
        float f_ = sigmoidf(fg);
        float g_ = tanhf(gg);
        float o_ = sigmoidf(og);
        float c  = f_ * cst[b][j] + i_ * g_;
        float h  = o_ * tanhf(c);
        cst[b][j] = c;
        hdst[(size_t)b * HID + hidx] = f2bf(h);
        if (l == 0) {
          out0[((size_t)t * BATCH + b) * HID + hidx] = f2bf(h);
        } else {
          out[((size_t)b * T_STEPS + t) * HID + hidx] = h;
        }
        if (t == T_STEPS - 1) {
          out[HN_BASE + (size_t)l * BATCH * HID + (size_t)b * HID + hidx] = h;
          out[CN_BASE + (size_t)l * BATCH * HID + (size_t)b * HID + hidx] = c;
        }
      }
    }

    // -------- device-scope step barrier (per layer, monotonic) --------
    __threadfence();
    __syncthreads();
    if (tid == 0) {
      __hip_atomic_fetch_add(&ctr[l], 1u, __ATOMIC_RELEASE,
                             __HIP_MEMORY_SCOPE_AGENT);
      unsigned target = (unsigned)(NWG * (t + 1));
      while (__hip_atomic_load(&ctr[l], __ATOMIC_ACQUIRE,
                               __HIP_MEMORY_SCOPE_AGENT) < target) {
        __builtin_amdgcn_s_sleep(1);
      }
    }
    __syncthreads();
  }
}

// ---------------------------------------------------------------------------
// Host launcher
// ---------------------------------------------------------------------------
extern "C" void kernel_launch(void* const* d_in, const int* in_sizes, int n_in,
                              void* d_out, int out_size, void* d_ws, size_t ws_size,
                              hipStream_t stream) {
  const float* x     = (const float*)d_in[0];
  const float* Wih0f = (const float*)d_in[1];
  const float* Whh0f = (const float*)d_in[2];
  const float* bih0  = (const float*)d_in[3];
  const float* bhh0  = (const float*)d_in[4];
  const float* Wih1f = (const float*)d_in[5];
  const float* Whh1f = (const float*)d_in[6];
  const float* bih1  = (const float*)d_in[7];
  const float* bhh1  = (const float*)d_in[8];
  float* out = (float*)d_out;

  char* ws = (char*)d_ws;
  const size_t WELEM = (size_t)4 * HID * 512;            // 1,048,576 per matrix
  unsigned short* wih0 = (unsigned short*)(ws + 0);
  unsigned short* whh0 = (unsigned short*)(ws + 2u * 1024 * 1024);
  unsigned short* wih1 = (unsigned short*)(ws + 4u * 1024 * 1024);
  unsigned short* whh1 = (unsigned short*)(ws + 6u * 1024 * 1024);
  unsigned short* hbuf = (unsigned short*)(ws + 8u * 1024 * 1024);   // 256 KB
  unsigned int*   ctr  = (unsigned int*)  (ws + 8u * 1024 * 1024 + 512u * 1024);
  unsigned short* out0 = (unsigned short*)(ws + 16u * 1024 * 1024);  // 128 MB

  init_ws<<<512, 256, 0, stream>>>(hbuf, ctr);
  cvt_bf16<<<(int)(WELEM / 1024), 256, 0, stream>>>(Wih0f, wih0, (int)WELEM);
  cvt_bf16<<<(int)(WELEM / 1024), 256, 0, stream>>>(Whh0f, whh0, (int)WELEM);
  cvt_bf16<<<(int)(WELEM / 1024), 256, 0, stream>>>(Wih1f, wih1, (int)WELEM);
  cvt_bf16<<<(int)(WELEM / 1024), 256, 0, stream>>>(Whh1f, whh1, (int)WELEM);

  lstm_persistent<<<2 * NWG, 256, 0, stream>>>(
      x, wih0, whh0, wih1, whh1, bih0, bhh0, bih1, bhh1,
      hbuf, out0, out, ctr);
}
